// SAD_64759516890049
// MI455X (gfx1250) — matrix operand order repaired and checked by reading, not compiled, not run on hardware
//
#include <hip/hip_runtime.h>
#include <math.h>

// ---------------------------------------------------------------------------
// Types for CDNA5 WMMA
// ---------------------------------------------------------------------------
typedef __attribute__((ext_vector_type(16))) _Float16 v16h;
typedef __attribute__((ext_vector_type(8)))  _Float16 v8h;
typedef __attribute__((ext_vector_type(8)))  float    v8f;

#define BN_EPS 1e-5f
#define POOL_EPS 1e-9f

// LDS tile geometry: 64 rows x 64 k-halves, row stride 72 halves (144 B,
// multiple of 16 B so every 8/16-half sub-run is 16B-aligned).
#define LDSK 72

// ---------------------------------------------------------------------------
// Wave macro-tile: 32x32 output (2x2 fragments) from LDS tiles, one 32-k slab
// starting at column `ks`.
// A lane fragment (16 halves): K = {hi*8..+7} ++ {16+hi*8..+7}   (row-major As)
// B lane fragment (16 halves): K = {hi*8..+7} ++ {16+hi*8..+7}, N = lane%16 (row-major Bt)
// ---------------------------------------------------------------------------
__device__ __forceinline__ void wmma_macro_tile(
    const _Float16 (*As)[LDSK], const _Float16 (*Bt)[LDSK],
    int wm, int wn, int lrow, int hi, int ks, v8f acc[2][2])
{
  v8h a0l = *(const v8h*)&As[wm + lrow][ks + hi * 8];
  v8h a0h = *(const v8h*)&As[wm + lrow][ks + 16 + hi * 8];
  v8h a1l = *(const v8h*)&As[wm + 16 + lrow][ks + hi * 8];
  v8h a1h = *(const v8h*)&As[wm + 16 + lrow][ks + 16 + hi * 8];
  v8h b0l = *(const v8h*)&Bt[wn + lrow][ks + hi * 8];
  v8h b0h = *(const v8h*)&Bt[wn + lrow][ks + 16 + hi * 8];
  v8h b1l = *(const v8h*)&Bt[wn + 16 + lrow][ks + hi * 8];
  v8h b1h = *(const v8h*)&Bt[wn + 16 + lrow][ks + 16 + hi * 8];

  v16h a0 = __builtin_shufflevector(a0l, a0h, 0,1,2,3,4,5,6,7,8,9,10,11,12,13,14,15);
  v16h a1 = __builtin_shufflevector(a1l, a1h, 0,1,2,3,4,5,6,7,8,9,10,11,12,13,14,15);
  v16h b0 = __builtin_shufflevector(b0l, b0h, 0,1,2,3,4,5,6,7,8,9,10,11,12,13,14,15);
  v16h b1 = __builtin_shufflevector(b1l, b1h, 0,1,2,3,4,5,6,7,8,9,10,11,12,13,14,15);

  acc[0][0] = __builtin_amdgcn_wmma_f32_16x16x32_f16(false, a0, false, b0, (short)0, acc[0][0], false, false);
  acc[0][1] = __builtin_amdgcn_wmma_f32_16x16x32_f16(false, a0, false, b1, (short)0, acc[0][1], false, false);
  acc[1][0] = __builtin_amdgcn_wmma_f32_16x16x32_f16(false, a1, false, b0, (short)0, acc[1][0], false, false);
  acc[1][1] = __builtin_amdgcn_wmma_f32_16x16x32_f16(false, a1, false, b1, (short)0, acc[1][1], false, false);
}

// ===========================================================================
// Implicit-GEMM convolution, WMMA f32_16x16x32_f16.
// GEMM view: M = OC, N = B*OH*OW, K = IC*KK*KK.
// Flattened weight index == oc*K + gk, so the A-tile is a plain matrix load.
// Block 128 threads / 4 waves; block tile 64x64; K-step 64.
// ===========================================================================
template<int KK>
__global__ __launch_bounds__(128) void conv_wmma(
    const float* __restrict__ X, const float* __restrict__ Wt,
    float* __restrict__ Y,
    int B, int IC, int IH, int IW,
    int OC, int OH, int OW,
    int stride, int pad)
{
  constexpr int KK2 = KK * KK;
  const int K   = IC * KK2;
  const int N   = B * OH * OW;
  const int OHW = OH * OW;
  const int IHW = IH * IW;

  __shared__ __align__(16) _Float16 As[64][LDSK];
  __shared__ __align__(16) _Float16 Bt[64][LDSK];

  const int tid  = threadIdx.x;
  const int wave = tid >> 5;
  const int lane = tid & 31;
  const int lrow = lane & 15;
  const int hi   = lane >> 4;

  const int m0 = blockIdx.y * 64;
  const int n0 = blockIdx.x * 64;
  const int wm = (wave >> 1) * 32;
  const int wn = (wave & 1) * 32;

  // A fill: fixed row m, 32 consecutive k per thread
  const int am  = tid & 63;
  const int akc = (tid >> 6) * 32;
  const int aoc = m0 + am;
  // B fill: fixed output pixel n, 32 consecutive k per thread
  const int bn  = tid & 63;
  const int bkc = (tid >> 6) * 32;
  const int gnB = n0 + bn;
  int bb = 0, boy = 0, box = 0;
  if (gnB < N) {
    bb = gnB / OHW; int r = gnB - bb * OHW;
    boy = r / OW;   box = r - boy * OW;
  }
  const int iy0 = boy * stride - pad;
  const int ix0 = box * stride - pad;
  // whole KKxKK window in-bounds? (uniform for entire kernel)
  const bool interior = (gnB < N) && (iy0 >= 0) && (iy0 + KK <= IH) &&
                        (ix0 >= 0) && (ix0 + KK <= IW);
  const float* __restrict__ Xbase = X + (size_t)bb * IC * IHW + iy0 * IW + ix0;
  const float* __restrict__ Wrow  = Wt + (size_t)aoc * K;

  v8f acc[2][2] = {{{}, {}}, {{}, {}}};

  for (int k0 = 0; k0 < K; k0 += 64) {
    __syncthreads();
    // ---------------- A tile ----------------
    {
      const int gkb = k0 + akc;
      if (aoc < OC && gkb + 32 <= K) {
        #pragma unroll
        for (int c = 0; c < 4; ++c) {
          v8h hv;
          #pragma unroll
          for (int e = 0; e < 8; ++e) hv[e] = (_Float16)Wrow[gkb + c * 8 + e];
          *(v8h*)&As[am][akc + c * 8] = hv;
        }
      } else {
        // branch-free rolled fallback: clamped load + select
        #pragma unroll 1
        for (int it = 0; it < 32; ++it) {
          int gk = gkb + it;
          bool ok = (aoc < OC) && (gk < K);
          size_t idx = ok ? ((size_t)aoc * K + gk) : 0;
          float v = Wt[idx];
          As[am][akc + it] = (_Float16)(ok ? v : 0.f);
        }
      }
    }
    // ---------------- B tile (im2col) ----------------
    {
      const int gkb = k0 + bkc;
      if (interior && gkb + 32 <= K) {
        #pragma unroll
        for (int c = 0; c < 4; ++c) {
          v8h hv;
          #pragma unroll
          for (int e = 0; e < 8; ++e) {
            int gk = gkb + c * 8 + e;
            int ic = gk / KK2; int rr = gk - ic * KK2;   // constant divisors
            int ky = rr / KK;  int kx = rr - ky * KK;
            hv[e] = (_Float16)Xbase[ic * IHW + ky * IW + kx];
          }
          *(v8h*)&Bt[bn][bkc + c * 8] = hv;
        }
      } else {
        // branch-free rolled fallback: clamped load + select
        #pragma unroll 1
        for (int it = 0; it < 32; ++it) {
          int gk = gkb + it;
          int ic = gk / KK2; int rr = gk - ic * KK2;
          int ky = rr / KK;  int kx = rr - ky * KK;
          int iy = iy0 + ky, ix = ix0 + kx;
          bool ok = (gnB < N) && (gk < K) &&
                    (iy >= 0) && (iy < IH) && (ix >= 0) && (ix < IW);
          size_t idx = ok ? ((size_t)bb * IC * IHW + (size_t)ic * IHW +
                             (size_t)iy * IW + ix) : 0;
          float v = X[idx];
          Bt[bn][bkc + it] = (_Float16)(ok ? v : 0.f);
        }
      }
    }
    __syncthreads();
    wmma_macro_tile(As, Bt, wm, wn, lrow, hi, 0,  acc);
    wmma_macro_tile(As, Bt, wm, wn, lrow, hi, 32, acc);
  }

  // ---- store: C VGPR r -> M = r + 8*hi, N = lane%16 ----
  #pragma unroll
  for (int j = 0; j < 2; ++j) {
    int gn = n0 + wn + j * 16 + lrow;
    if (gn >= N) continue;
    int b = gn / OHW; int r = gn - b * OHW;
    int oy = r / OW;  int ox = r - oy * OW;
    float* __restrict__ Yb = Y + (size_t)b * OC * OHW + oy * OW + ox;
    #pragma unroll
    for (int i = 0; i < 2; ++i) {
      int ocb = m0 + wm + i * 16 + 8 * hi;
      if (ocb + 8 <= OC) {
        #pragma unroll
        for (int rI = 0; rI < 8; ++rI)
          Yb[(size_t)(ocb + rI) * OHW] = acc[i][j][rI];
      } else {
        #pragma unroll 1
        for (int rI = 0; rI < 8; ++rI)
          if (ocb + rI < OC) Yb[(size_t)(ocb + rI) * OHW] = acc[i][j][rI];
      }
    }
  }
}

// ===========================================================================
// Generic WMMA GEMM: C[MxN] = A[MxK] * Bm[KxN] (+ bias0[m] + bias1[m])
// Same 64x64 block tile, K-step 64.
// ===========================================================================
__global__ __launch_bounds__(128) void gemm_wmma(
    const float* __restrict__ A, const float* __restrict__ Bm,
    float* __restrict__ C, int M, int N, int K,
    const float* __restrict__ bias0, const float* __restrict__ bias1)
{
  __shared__ __align__(16) _Float16 As[64][LDSK];
  __shared__ __align__(16) _Float16 Bt[64][LDSK];

  const int tid  = threadIdx.x;
  const int wave = tid >> 5;
  const int lane = tid & 31;
  const int lrow = lane & 15;
  const int hi   = lane >> 4;

  const int m0 = blockIdx.y * 64;
  const int n0 = blockIdx.x * 64;
  const int wm = (wave >> 1) * 32;
  const int wn = (wave & 1) * 32;

  // A fill: fixed row, 32 consecutive k
  const int am  = tid & 63;
  const int akc = (tid >> 6) * 32;
  const int agm = m0 + am;
  // B fill: fixed k, 32 consecutive n (contiguous global loads)
  const int bkf = tid & 63;
  const int bnc = (tid >> 6) * 32;

  v8f acc[2][2] = {{{}, {}}, {{}, {}}};

  for (int k0 = 0; k0 < K; k0 += 64) {
    __syncthreads();
    {
      const int gkb = k0 + akc;
      const float* __restrict__ Arow = A + (size_t)agm * K;
      if (agm < M && gkb + 32 <= K) {
        #pragma unroll
        for (int c = 0; c < 4; ++c) {
          v8h hv;
          #pragma unroll
          for (int e = 0; e < 8; ++e) hv[e] = (_Float16)Arow[gkb + c * 8 + e];
          *(v8h*)&As[am][akc + c * 8] = hv;
        }
      } else {
        #pragma unroll 1
        for (int it = 0; it < 32; ++it) {
          int gk = gkb + it;
          bool ok = (agm < M) && (gk < K);
          size_t idx = ok ? ((size_t)agm * K + gk) : 0;
          float v = A[idx];
          As[am][akc + it] = (_Float16)(ok ? v : 0.f);
        }
      }
    }
    {
      const int gk  = k0 + bkf;
      const int gnb = n0 + bnc;
      if (gk < K && gnb + 32 <= N) {
        const float* __restrict__ Brow = Bm + (size_t)gk * N + gnb;
        #pragma unroll
        for (int it = 0; it < 32; ++it)
          Bt[bnc + it][bkf] = (_Float16)Brow[it];
      } else {
        #pragma unroll 1
        for (int it = 0; it < 32; ++it) {
          int gn = gnb + it;
          bool ok = (gk < K) && (gn < N);
          size_t idx = ok ? ((size_t)gk * N + gn) : 0;
          float v = Bm[idx];
          Bt[bnc + it][bkf] = (_Float16)(ok ? v : 0.f);
        }
      }
    }
    __syncthreads();
    wmma_macro_tile(As, Bt, wm, wn, lrow, hi, 0,  acc);
    wmma_macro_tile(As, Bt, wm, wn, lrow, hi, 32, acc);
  }

  #pragma unroll
  for (int j = 0; j < 2; ++j) {
    int gn = n0 + wn + j * 16 + lrow;
    if (gn >= N) continue;
    #pragma unroll
    for (int i = 0; i < 2; ++i) {
      int gmb = m0 + wm + i * 16 + 8 * hi;
      if (gmb + 8 <= M) {
        #pragma unroll
        for (int rI = 0; rI < 8; ++rI) {
          int gm = gmb + rI;
          float v = acc[i][j][rI];
          if (bias0) v += bias0[gm];
          if (bias1) v += bias1[gm];
          C[(size_t)gm * N + gn] = v;
        }
      } else {
        #pragma unroll 1
        for (int rI = 0; rI < 8; ++rI) {
          int gm = gmb + rI;
          if (gm < M) {
            float v = acc[i][j][rI];
            if (bias0) v += bias0[gm];
            if (bias1) v += bias1[gm];
            C[(size_t)gm * N + gn] = v;
          }
        }
      }
    }
  }
}

// ===========================================================================
// Elementwise epilogues
// ===========================================================================
__global__ void ew_post_conv1(float* X, const float* bias,
                              const float* g, const float* be,
                              const float* me, const float* va,
                              int C, int HW, int total)
{
  int i = blockIdx.x * blockDim.x + threadIdx.x;
  if (i >= total) return;
  int c = (i / HW) % C;
  float x = X[i] + bias[c];
  x = fmaxf(x, 0.f);
  float sc = g[c] * rsqrtf(va[c] + BN_EPS);
  X[i] = (x - me[c]) * sc + be[c];
}

__global__ void ew_bn(float* X, const float* g, const float* be,
                      const float* me, const float* va,
                      int C, int HW, int total, int relu)
{
  int i = blockIdx.x * blockDim.x + threadIdx.x;
  if (i >= total) return;
  int c = (i / HW) % C;
  float sc = g[c] * rsqrtf(va[c] + BN_EPS);
  float x = (X[i] - me[c]) * sc + be[c];
  X[i] = relu ? fmaxf(x, 0.f) : x;
}

__global__ void ew_add_relu(float* X, const float* __restrict__ I, int total)
{
  int i = blockIdx.x * blockDim.x + threadIdx.x;
  if (i >= total) return;
  X[i] = fmaxf(X[i] + I[i], 0.f);
}

// ===========================================================================
// Stats pooling: input (B,C,T,F) NCHW. Feature u = c*F+f, column n = b*T+t.
// Window-5 reflect-padded mean/std over time -> P[(u | CF+u)][n], (2CF x BT)
// ===========================================================================
__global__ void stats_pool(const float* __restrict__ X, float* __restrict__ P,
                           int Bn, int C, int T, int F)
{
  int NT = Bn * T;
  int CF = C * F;
  int idx = blockIdx.x * blockDim.x + threadIdx.x;
  if (idx >= CF * NT) return;
  int u = idx / NT;  int n = idx - u * NT;
  int b = n / T;     int t = n - b * T;
  int c = u / F;     int f = u - c * F;
  float s = 0.f, s2 = 0.f;
  #pragma unroll
  for (int d = -2; d <= 2; ++d) {
    int tt = t + d;
    if (tt < 0) tt = -tt;
    if (tt >= T) tt = 2 * T - 2 - tt;
    float v = X[((b * C + c) * T + tt) * F + f];
    s += v; s2 += v * v;
  }
  float mean = s * 0.2f;
  float var  = fmaxf(s2 * 0.2f - mean * mean, POOL_EPS);
  P[u * NT + n]        = mean;
  P[(CF + u) * NT + n] = sqrtf(var);
}

__global__ void l2norm_cols(float* X, int M, int N)
{
  int n = blockIdx.x;
  int tid = threadIdx.x;
  __shared__ float red[128];
  float v = X[tid * N + n];
  red[tid] = v * v;
  __syncthreads();
  for (int s = 64; s > 0; s >>= 1) {
    if (tid < s) red[tid] += red[tid + s];
    __syncthreads();
  }
  float nrm = fmaxf(sqrtf(red[0]), 1e-12f);
  X[tid * N + n] = v / nrm;
}

// ===========================================================================
// LSTM recurrence; one block per batch element, 512 threads = one gate each.
// Gate order (jnp.split): i, f, g, o.
// ===========================================================================
__global__ __launch_bounds__(512) void lstm_scan(
    const float* __restrict__ XW, const float* __restrict__ Whh,
    float* __restrict__ Hout, int N, int T, int reverse, int rowOff, int ldH)
{
  int b = blockIdx.x;
  int g = threadIdx.x;
  __shared__ float h[128], c[128], gates[512];
  if (g < 128) { h[g] = 0.f; c[g] = 0.f; }
  __syncthreads();

  const float* wr = Whh + g * 128;
  for (int s = 0; s < T; ++s) {
    int t = reverse ? (T - 1 - s) : s;
    int n = b * T + t;
    float acc = XW[g * N + n];
    #pragma unroll 8
    for (int j = 0; j < 128; ++j) acc += wr[j] * h[j];
    gates[g] = acc;
    __syncthreads();
    if (g < 128) {
      float ig = 1.f / (1.f + __expf(-gates[g]));
      float fg = 1.f / (1.f + __expf(-gates[128 + g]));
      float gg = tanhf(gates[256 + g]);
      float og = 1.f / (1.f + __expf(-gates[384 + g]));
      float cn = fg * c[g] + ig * gg;
      float hn = og * tanhf(cn);
      c[g] = cn; h[g] = hn;
      Hout[(rowOff + g) * ldH + n] = hn;
    }
    __syncthreads();
  }
}

__global__ void out_proj(const float* __restrict__ H, const float* __restrict__ ow,
                         const float* __restrict__ ob, float* __restrict__ out, int N)
{
  int n = blockIdx.x * blockDim.x + threadIdx.x;
  if (n >= N) return;
  float acc = ob[0];
  #pragma unroll 8
  for (int j = 0; j < 256; ++j) acc += H[j * N + n] * ow[j];
  out[n] = acc;
}

// ===========================================================================
// Host orchestration
// ===========================================================================
struct BNp { const float *g, *b, *m, *v; };
struct Blk { const float* c1w; BNp b1; const float* c2w; BNp b2;
             const float* dw; BNp dbn; bool down; };
struct Dir { const float *Wih, *Whh, *bih, *bhh; };

extern "C" void kernel_launch(void* const* d_in, const int* in_sizes, int n_in,
                              void* d_out, int out_size, void* d_ws, size_t ws_size,
                              hipStream_t stream)
{
  (void)in_sizes; (void)n_in; (void)out_size; (void)ws_size;

  // ---- walk d_in in setup_inputs() insertion order ----
  int cur = 0;
  auto nx = [&]() -> const float* { return (const float*)d_in[cur++]; };
  auto nbn = [&]() -> BNp { BNp p; p.g = nx(); p.b = nx(); p.m = nx(); p.v = nx(); return p; };

  const float* feats   = nx();
  const float* conv1_w = nx();
  const float* conv1_b = nx();
  BNp bn1 = nbn();

  static const int NF[4]  = {32, 64, 128, 256};
  static const int NL[4]  = {3, 4, 6, 3};
  static const int STR[4] = {1, 2, 2, 2};

  Blk blocks[4][6];
  int inpl = 32;
  for (int li = 0; li < 4; ++li) {
    for (int bi = 0; bi < NL[li]; ++bi) {
      Blk& bk = blocks[li][bi];
      bk.c1w = nx(); bk.b1 = nbn();
      bk.c2w = nx(); bk.b2 = nbn();
      bk.down = (bi == 0) && (STR[li] != 1 || inpl != NF[li]);
      if (bk.down) { bk.dw = nx(); bk.dbn = nbn(); }
      else         { bk.dw = nullptr; }
      inpl = NF[li];
    }
  }

  const float* fc_w = nx();
  const float* fc_b = nx();

  Dir dirs[4][2];   // sd0, sd1, cb0, cb1 ; [0]=fwd, [1]=bwd
  for (int L = 0; L < 4; ++L)
    for (int d = 0; d < 2; ++d) {
      Dir& dd = dirs[L][d];
      dd.Wih = nx(); dd.Whh = nx(); dd.bih = nx(); dd.bhh = nx();
    }
  const float* out_w = nx();
  const float* out_b = nx();

  // ---- workspace layout (floats) ----
  float* ws = (float*)d_ws;
  const size_t BUFE = (size_t)4 * 32 * 1600 * 80;   // 16,384,000 floats
  float* Xb = ws;
  float* S1 = ws + BUFE;
  float* S2 = ws + 2 * BUFE;
  float* S3 = ws + 3 * BUFE;
  float* P   = ws + 4 * BUFE;            // 5120 x 800
  float* FC  = P + (size_t)5120 * 800;   // 128 x 800
  float* XW0 = FC + (size_t)128 * 800;   // 512 x 800
  float* XW1 = XW0 + (size_t)512 * 800;  // 512 x 800
  float* HA  = XW1 + (size_t)512 * 800;  // 256 x 800
  float* HB  = HA + (size_t)256 * 800;   // 256 x 800

  const int B = 4;

  auto conv = [&](const float* X, const float* W, float* Y,
                  int IC, int IH, int IW, int OC, int OH, int OW,
                  int kk, int st, int pd) {
    int N = B * OH * OW;
    dim3 grid((N + 63) / 64, (OC + 63) / 64);
    if (kk == 3)
      conv_wmma<3><<<grid, 128, 0, stream>>>(X, W, Y, B, IC, IH, IW, OC, OH, OW, st, pd);
    else
      conv_wmma<1><<<grid, 128, 0, stream>>>(X, W, Y, B, IC, IH, IW, OC, OH, OW, st, pd);
  };
  auto gemm = [&](const float* A, const float* Bm, float* C,
                  int M, int N, int K, const float* b0, const float* b1) {
    dim3 grid((N + 63) / 64, (M + 63) / 64);
    gemm_wmma<<<grid, 128, 0, stream>>>(A, Bm, C, M, N, K, b0, b1);
  };

  // ---- stem: conv1 + bias + relu + bn ----
  int C = 32, H = 1600, Wd = 80;
  conv(feats, conv1_w, Xb, 1, 1600, 80, 32, 1600, 80, 3, 1, 1);
  {
    int total = B * C * H * Wd;
    ew_post_conv1<<<(total + 255) / 256, 256, 0, stream>>>(
        Xb, conv1_b, bn1.g, bn1.b, bn1.m, bn1.v, C, H * Wd, total);
  }

  // ---- residual layers ----
  for (int li = 0; li < 4; ++li) {
    int planes = NF[li];
    for (int bi = 0; bi < NL[li]; ++bi) {
      const Blk& bk = blocks[li][bi];
      int s  = (bi == 0) ? STR[li] : 1;
      int OH = (H + 2 - 3) / s + 1;
      int OW = (Wd + 2 - 3) / s + 1;
      int total = B * planes * OH * OW;

      conv(Xb, bk.c1w, S1, C, H, Wd, planes, OH, OW, 3, s, 1);
      ew_bn<<<(total + 255) / 256, 256, 0, stream>>>(
          S1, bk.b1.g, bk.b1.b, bk.b1.m, bk.b1.v, planes, OH * OW, total, 1);

      conv(S1, bk.c2w, S2, planes, OH, OW, planes, OH, OW, 3, 1, 1);
      ew_bn<<<(total + 255) / 256, 256, 0, stream>>>(
          S2, bk.b2.g, bk.b2.b, bk.b2.m, bk.b2.v, planes, OH * OW, total, 0);

      const float* idn;
      if (bk.down) {
        conv(Xb, bk.dw, S3, C, H, Wd, planes, OH, OW, 1, s, 0);
        ew_bn<<<(total + 255) / 256, 256, 0, stream>>>(
            S3, bk.dbn.g, bk.dbn.b, bk.dbn.m, bk.dbn.v, planes, OH * OW, total, 0);
        idn = S3;
      } else {
        idn = Xb;
      }
      ew_add_relu<<<(total + 255) / 256, 256, 0, stream>>>(S2, idn, total);

      float* tmp = Xb; Xb = S2; S2 = tmp;
      C = planes; H = OH; Wd = OW;
    }
  }
  // Xb now (4, 256, 200, 10)

  // ---- stats pooling -> P (5120 x 800, feature-major) ----
  {
    int total = 256 * 10 * B * 200;
    stats_pool<<<(total + 255) / 256, 256, 0, stream>>>(Xb, P, B, 256, 200, 10);
  }

  // ---- FC + L2 norm ----
  gemm(fc_w, P, FC, 128, 800, 5120, fc_b, nullptr);
  l2norm_cols<<<800, 128, 0, stream>>>(FC, 128, 800);

  // ---- 4 stacked BiLSTM layers (sd0, sd1, cb0, cb1) ----
  const float* xin = FC; int inF = 128;
  float* hout = HA;
  for (int L = 0; L < 4; ++L) {
    const Dir& f  = dirs[L][0];
    const Dir& bw = dirs[L][1];
    gemm(f.Wih,  xin, XW0, 512, 800, inF, f.bih,  f.bhh);
    gemm(bw.Wih, xin, XW1, 512, 800, inF, bw.bih, bw.bhh);
    lstm_scan<<<4, 512, 0, stream>>>(XW0, f.Whh,  hout, 800, 200, 0, 0,   800);
    lstm_scan<<<4, 512, 0, stream>>>(XW1, bw.Whh, hout, 800, 200, 1, 128, 800);
    xin = hout; inF = 256;
    hout = (hout == HA) ? HB : HA;
  }

  // ---- output projection -> d_out (4,200,1) = 800 floats ----
  out_proj<<<(800 + 255) / 256, 256, 0, stream>>>(
      xin, out_w, out_b, (float*)d_out, 800);
}
